// Encoder_4758823764201
// MI455X (gfx1250) — compile-verified, matrix operand order checked by reading, not run on hardware
//
#include <hip/hip_runtime.h>
#include <stdint.h>

typedef __attribute__((ext_vector_type(8))) int v8i;

#define B_    128
#define S_    128
#define V_    10000
#define D_    10000
#define WPR   313     /* 32-bit packed words per row (313*32 >= 10000) */
#define STRIDE 320    /* u32 stride per packed row: 313 words + wrap word @313, padded */
#define NW    126     /* number of trigram windows = S - 2 */

/* ---------------- pass 0: mark referenced embedding rows ---------------- */
__global__ void mark_used(const int* __restrict__ x, unsigned* __restrict__ used) {
    int t = threadIdx.x;
    for (int i = t; i < V_; i += blockDim.x) used[i] = 0u;
    __syncthreads();
    for (int i = t; i < B_ * S_; i += blockDim.x) used[x[i]] = 1u;  // same-value races OK
}

/* ---------------- pass 1: bit-pack sign bits of used rows ----------------
   P[v*STRIDE + w], w in [0,312]: bit i = sign(E[v][32w+i]); word 312 has 16 valid bits,
   high bits forced to 0. P[v*STRIDE + 313] = wrap word: bit i = sign(E[v][9968+i]),
   i.e. the word holding bits d-2, d-1 for d in {0,1} of the circular roll.          */
__global__ void pack_rows(const unsigned* __restrict__ Eu,
                          const unsigned* __restrict__ used,
                          unsigned* __restrict__ P) {
    int v = blockIdx.y;
    if (!used[v]) return;
    int w = blockIdx.x * blockDim.x + threadIdx.x;
    if (w > WPR) return;
    int base = (w == WPR) ? (D_ - 32) : (w * 32);
    const unsigned* row = Eu + (size_t)v * D_;
    unsigned word = 0;
#pragma unroll
    for (int i = 0; i < 8; ++i) {
        int o = base + 4 * i;
        if (o + 3 < D_) {                       /* D_ % 16 == 0, so all-or-nothing */
            uint4 q = *(const uint4*)(row + o); /* 16B aligned: rows are 40000B   */
            word |= (q.x >> 31) << (4 * i);
            word |= (q.y >> 31) << (4 * i + 1);
            word |= (q.z >> 31) << (4 * i + 2);
            word |= (q.w >> 31) << (4 * i + 3);
        }
    }
    P[(size_t)v * STRIDE + w] = word;
}

/* ---------------- pass 2: XOR + bit-sliced count + WMMA reduction --------- */
__global__ __launch_bounds__(64) void encode_wmma(const int* __restrict__ x,
                                                  const unsigned* __restrict__ P,
                                                  float* __restrict__ out) {
    __shared__ int xs[S_];
    int b   = blockIdx.y;
    int tid = threadIdx.x;
    for (int i = tid; i < S_; i += 64) xs[i] = x[b * S_ + i];
    __syncthreads();

    int lane   = tid & 31;
    int waveId = tid >> 5;
    int wbase  = blockIdx.x * 64 + waveId * 32;      /* 0,32,...,288 */
    int w = wbase + lane; if (w > 312) w = 312;      /* clamp lanes past last word  */
    int wprev = (w == 0) ? WPR : (w - 1);            /* wrap word handles d-1,d-2<0 */

    /* bit-plane counters: cnt(d) = sum_i 2^i * bit_i, max 126 fits 7 planes.
       Rolling register reuse: window j's rows are windows j+1/j+2's rows, so only
       row x[j+2] needs fresh loads (2 per window instead of 5).                   */
    unsigned c0 = 0, c1 = 0, c2 = 0, c3 = 0, c4 = 0, c5 = 0, c6 = 0;
    {
        const unsigned* q0 = P + (size_t)xs[0] * STRIDE;
        const unsigned* q1 = P + (size_t)xs[1] * STRIDE;
        unsigned h0 = q0[w], l0 = q0[wprev];         /* factor rolled by 2 */
        unsigned h1 = q1[w], l1 = q1[wprev];         /* factor rolled by 1 */
        for (int j = 0; j < NW; ++j) {
            const unsigned* q2 = P + (size_t)xs[j + 2] * STRIDE;
            unsigned h2 = q2[w], l2 = q2[wprev];
            /* bit i of t: sign of the triple product at d = 32w + i */
            unsigned t = ((h0 << 2) | (l0 >> 30)) ^ ((h1 << 1) | (l1 >> 31)) ^ h2;
            /* carry-save add of one bit per dim into 7 planes */
            unsigned xx = t, cr;
            cr = c0 & xx; c0 ^= xx; xx = cr;
            cr = c1 & xx; c1 ^= xx; xx = cr;
            cr = c2 & xx; c2 ^= xx; xx = cr;
            cr = c3 & xx; c3 ^= xx; xx = cr;
            cr = c4 & xx; c4 ^= xx; xx = cr;
            cr = c5 & xx; c5 ^= xx; xx = cr;
            c6 ^= xx;
            h0 = h1; l0 = l1; h1 = h2; l1 = l2;
        }
    }

    /* Epilogue: cnt = sum_i 2^i*bit_i via V_WMMA_I32_16X16X64_IU8 with B = ones.
       A row m <- lane m (tile0) or lane m+16 (tile1); bytes are bit_i * 2^i placed
       in 7 of the lane's K-slots. Since B is all-ones, D[m,n] = sum of row-m bytes,
       independent of the K-slot permutation inside a lane.                         */
    v8i ones;
#pragma unroll
    for (int i = 0; i < 8; ++i) ones[i] = 0x01010101;
    bool hiHalf = lane >= 16;
    int  laneN  = lane & 15;
    float* outb = out + (size_t)b * D_;

    /* Branch-free writeback: every lane writes exactly one (tile,row):
         laneN 0..7  -> tile0, row M = laneN     + 8*hiHalf
         laneN 8..15 -> tile1, row M = (laneN&7) + 8*hiHalf
       Lane L holds D[M = r + 8*hiHalf, N = laneN] in C/D VGPR r; columns are
       identical (B = ones), so D[rsel] of the chosen tile is our row's count.      */
    int  rsel = laneN & 7;
    int  tsel = laneN >> 3;
    int  wout = wbase + 16 * tsel + rsel + (hiHalf ? 8 : 0);
    if (wout > 312) wout = 312;                    /* dup writes of identical value */
    float* wrbase = outb + 32 * wout;
    int    dlim   = D_ - 32 * wout;                /* store iff dp < dlim */

#define BUILD_TILES(DP, RD0, RD1)                                               \
    {                                                                           \
        unsigned e0 = (c0 >> (DP)) & 1u, e1 = (c1 >> (DP)) & 1u;                \
        unsigned e2 = (c2 >> (DP)) & 1u, e3 = (c3 >> (DP)) & 1u;                \
        unsigned e4 = (c4 >> (DP)) & 1u, e5 = (c5 >> (DP)) & 1u;                \
        unsigned e6 = (c6 >> (DP)) & 1u;                                        \
        unsigned va0 = e0 | (e1 << 9) | (e2 << 18) | (e3 << 27);                \
        unsigned va1 = (e4 << 4) | (e5 << 13) | (e6 << 22);                     \
        v8i A0 = {0, 0, 0, 0, 0, 0, 0, 0};                                      \
        v8i A1 = {0, 0, 0, 0, 0, 0, 0, 0};                                      \
        A0[0] = hiHalf ? 0 : (int)va0;                                          \
        A0[1] = hiHalf ? 0 : (int)va1;                                          \
        A1[0] = hiHalf ? (int)va0 : 0;                                          \
        A1[1] = hiHalf ? (int)va1 : 0;                                          \
        v8i Cz = {0, 0, 0, 0, 0, 0, 0, 0};                                      \
        RD0 = __builtin_amdgcn_wmma_i32_16x16x64_iu8(false, A0, false, ones, Cz, false, false); \
        RD1 = __builtin_amdgcn_wmma_i32_16x16x64_iu8(false, A1, false, ones, Cz, false, false); \
    }

#define CONSUME_TILES(DP, RD0, RD1)                                             \
    {                                                                           \
        int val = tsel ? RD1[0] : RD0[0];                                       \
        _Pragma("unroll")                                                       \
        for (int r = 1; r < 8; ++r) {                                           \
            int cand = tsel ? RD1[r] : RD0[r];                                  \
            val = (rsel == r) ? cand : val;                                     \
        }                                                                       \
        if ((DP) < dlim) wrbase[(DP)] = (val <= 62) ? 1.0f : -1.0f;             \
    }

    /* Software-pipelined: consume dp-1 while dp's WMMAs are in flight, so the
       select/store VALU work fills the WMMA->VALU hazard slots instead of v_nops. */
    v8i pD0, pD1;
    BUILD_TILES(0, pD0, pD1);
    for (int dp = 1; dp < 32; ++dp) {
        v8i nD0, nD1;
        BUILD_TILES(dp, nD0, nD1);
        CONSUME_TILES(dp - 1, pD0, pD1);
        pD0 = nD0; pD1 = nD1;
    }
    CONSUME_TILES(31, pD0, pD1);
#undef BUILD_TILES
#undef CONSUME_TILES
}

/* ---------------- fallback: direct gather (used only if ws too small) ------ */
__global__ void encode_naive(const int* __restrict__ x, const float* __restrict__ E,
                             float* __restrict__ out) {
    int b = blockIdx.y;
    int d = blockIdx.x * blockDim.x + threadIdx.x;
    if (d >= D_) return;
    int dm1 = d - 1; if (dm1 < 0) dm1 += D_;
    int dm2 = d - 2; if (dm2 < 0) dm2 += D_;
    const int* xr = x + b * S_;
    float s = 0.f;
    int i0 = xr[0], i1 = xr[1];
    for (int j = 0; j < NW; ++j) {
        int i2 = xr[j + 2];
        s += E[(size_t)i0 * D_ + dm2] * E[(size_t)i1 * D_ + dm1] * E[(size_t)i2 * D_ + d];
        i0 = i1; i1 = i2;
    }
    out[(size_t)b * D_ + d] = (s > 0.f) ? 1.0f : -1.0f;
}

extern "C" void kernel_launch(void* const* d_in, const int* in_sizes, int n_in,
                              void* d_out, int out_size, void* d_ws, size_t ws_size,
                              hipStream_t stream) {
    const int*   x = (const int*)d_in[0];
    const float* E = (const float*)d_in[1];
    float*     out = (float*)d_out;

    const size_t usedBytes = 40960;                       /* 10000 flags, padded */
    const size_t packBytes = (size_t)V_ * STRIDE * 4;     /* 12.8 MB packed table */
    if (ws_size < usedBytes + packBytes) {
        dim3 g((D_ + 255) / 256, B_);
        encode_naive<<<g, 256, 0, stream>>>(x, E, out);
        return;
    }
    unsigned* used = (unsigned*)d_ws;
    unsigned* P    = (unsigned*)((char*)d_ws + usedBytes);

    mark_used<<<1, 1024, 0, stream>>>(x, used);
    dim3 g1(5, V_);                 /* 5*64 = 320 threads cover 314 words */
    pack_rows<<<g1, 64, 0, stream>>>((const unsigned*)E, used, P);
    dim3 g2(5, B_);                 /* 5 chunks of 64 words x 128 batches */
    encode_wmma<<<g2, 64, 0, stream>>>(x, P, out);
}
